// MultiheadAttention_3728031613550
// MI455X (gfx1250) — matrix operand order repaired and checked by reading, not compiled, not run on hardware
//
#include <hip/hip_runtime.h>
#include <stdint.h>

// Problem constants (B=2, S=2048, E=1024, H=16 -> BH=32 scrambled heads, Hd=64)
#define S_LEN 2048
#define EMB   1024
#define NROWS 4096            // B*S
#define NQ    4194304ull      // B*S*E elements
#define NW    1048576ull      // E*E elements

typedef __attribute__((ext_vector_type(16))) __bf16 v16bf;
typedef __attribute__((ext_vector_type(8)))  float  v8f;

union FragB {
  v16bf  v;
  uint4  q[2];
  __bf16 h[16];
};

// ---------------------------------------------------------------------------
// fp32 -> bf16 conversion (vectorized, n multiple of 1024 always)
// ---------------------------------------------------------------------------
__global__ void __launch_bounds__(256) cvt_f32_to_bf16(const float* __restrict__ src,
                                                       __bf16* __restrict__ dst, int n)
{
  int i = (blockIdx.x * 256 + threadIdx.x) * 4;
  if (i + 3 < n) {
    float4 f = *(const float4*)(src + i);
    dst[i + 0] = (__bf16)f.x;
    dst[i + 1] = (__bf16)f.y;
    dst[i + 2] = (__bf16)f.z;
    dst[i + 3] = (__bf16)f.w;
  }
}

// ---------------------------------------------------------------------------
// Y = relu(X (4096x1024, bf16) @ W^T (W is 1024x1024 row-major [out][in]) + b)
// mode 0: Y -> bf16 row-major       (q/k projections)
// mode 1: Y -> bf16 transposed-per-head vT[(bh*64+d)*2048 + s2]  (v projection)
// mode 2: Y -> fp32 row-major to d_out (output projection)
// Block: 256 threads (8 waves). WG tile 64x64, wave tile 16x32 (2 WMMA/k-step).
// Fragments read straight from global: per the CDNA5 16-bit A/B layouts every
// lane's slice is a contiguous 16B chunk -> global_load_b128, L2-resident.
// ---------------------------------------------------------------------------
__global__ void __launch_bounds__(256) gemm_relu_bf16(const __bf16* __restrict__ X,
                                                      const __bf16* __restrict__ W,
                                                      const float*  __restrict__ bias,
                                                      float*        __restrict__ outF,
                                                      __bf16*       __restrict__ outB,
                                                      int mode)
{
  const int wave = threadIdx.x >> 5;
  const int lane = threadIdx.x & 31;
  const int m    = lane & 15;
  const int half = lane >> 4;
  const int row0 = blockIdx.y * 64 + (wave >> 1) * 16;
  const int col0 = blockIdx.x * 64 + (wave & 1) * 32;

  const v8f vzero = {0.f,0.f,0.f,0.f,0.f,0.f,0.f,0.f};
  v8f acc0 = vzero, acc1 = vzero;

  const __bf16* xr = X + (size_t)(row0 + m) * EMB;
  const __bf16* w0 = W + (size_t)(col0 + m) * EMB;
  const __bf16* w1 = W + (size_t)(col0 + 16 + m) * EMB;

  for (int kb = 0; kb < EMB; kb += 32) {
    FragB a, b0, b1;
    // A 16x32 bf16: lanes 0-15 hold K[0..15]; lanes 16-31 K[16..31] (contiguous)
    a.q[0]  = *(const uint4*)(xr + kb + half * 16);
    a.q[1]  = *(const uint4*)(xr + kb + half * 16 + 8);
    // B 32x16 bf16: lane n=lane&15, elements e -> K = (lane>>4)*16 + e (contiguous)
    b0.q[0] = *(const uint4*)(w0 + kb + half * 16);
    b0.q[1] = *(const uint4*)(w0 + kb + half * 16 + 8);
    b1.q[0] = *(const uint4*)(w1 + kb + half * 16);
    b1.q[1] = *(const uint4*)(w1 + kb + half * 16 + 8);
    acc0 = __builtin_amdgcn_wmma_f32_16x16x32_bf16(false, a.v, false, b0.v, (short)0, acc0, false, false);
    acc1 = __builtin_amdgcn_wmma_f32_16x16x32_bf16(false, a.v, false, b1.v, (short)0, acc1, false, false);
  }

  // C/D layout: element r -> (M = r + 8*half, N = lane&15)
  #pragma unroll
  for (int r = 0; r < 8; ++r) {
    int row = row0 + r + half * 8;
    #pragma unroll
    for (int t = 0; t < 2; ++t) {
      int   col = col0 + t * 16 + m;
      float v   = (t == 0 ? acc0[r] : acc1[r]) + bias[col];
      v = fmaxf(v, 0.0f);
      if (mode == 2) {
        outF[(size_t)row * EMB + col] = v;
      } else if (mode == 0) {
        outB[(size_t)row * EMB + col] = (__bf16)v;
      } else {
        int flat = row * EMB + col;
        int s2 = flat >> 11, bh = (flat >> 6) & 31, d = flat & 63;
        outB[(size_t)(bh * 64 + d) * S_LEN + s2] = (__bf16)v;
      }
    }
  }
}

// ---------------------------------------------------------------------------
// Attention strip kernel: one WG = head bh (0..31) x 16-query strip s (0..127).
// LDS: P[16][2048] fp32 scores strip (128KB) + Q strip bf16 + reduction pad.
// ---------------------------------------------------------------------------
#define ATTN_SMEM (16 * 2048 * 4 + 16 * 64 * 2 + 16 * 16 * 4)

__global__ void __launch_bounds__(256) attn_strip(const __bf16* __restrict__ qp,
                                                  const __bf16* __restrict__ kp,
                                                  const __bf16* __restrict__ vT,
                                                  float*        __restrict__ aw,
                                                  __bf16*       __restrict__ x2)
{
  extern __shared__ char smem[];
  float*  P    = (float*)smem;                          // 16 x 2048 scores/probs
  __bf16* qlds = (__bf16*)(smem + 16 * 2048 * 4);       // 16 x 64 Q strip
  float*  red  = (float*)(smem + 16 * 2048 * 4 + 16 * 64 * 2); // 16 x 16 partials

  const int s    = blockIdx.x;
  const int bh   = blockIdx.y;
  const int tid  = threadIdx.x;
  const int wave = tid >> 5;
  const int lane = tid & 31;
  const int m    = lane & 15;
  const int half = lane >> 4;
  const v8f vzero = {0.f,0.f,0.f,0.f,0.f,0.f,0.f,0.f};

  // ---- load Q strip (16 rows x 64 bf16) into LDS ----
  {
    int row = tid >> 4, seg = tid & 15;
    const __bf16* src = qp + (size_t)(s * 16 + row) * 2048 + bh * 64 + seg * 4;
    *(unsigned long long*)(qlds + row * 64 + seg * 4) = *(const unsigned long long*)src;
  }
  __syncthreads();

  // ---- Q fragments (K=0..31 and K=32..63), held in registers ----
  FragB aq0, aq1;
  {
    const __bf16* qr = qlds + m * 64;
    aq0.q[0] = *(const uint4*)(qr + half * 16);
    aq0.q[1] = *(const uint4*)(qr + half * 16 + 8);
    aq1.q[0] = *(const uint4*)(qr + 32 + half * 16);
    aq1.q[1] = *(const uint4*)(qr + 40 + half * 16);
  }

  // ---- scores: Q(16x64) . K^T -> P strip; causal mask; tiles split by wave ----
  for (int jt = wave; jt < 128; jt += 8) {
    if (jt <= s) {
      FragB b0, b1;
      const __bf16* kr = kp + (size_t)(jt * 16 + m) * 2048 + bh * 64 + half * 16;
      b0.q[0] = *(const uint4*)(kr);
      b0.q[1] = *(const uint4*)(kr + 8);
      b1.q[0] = *(const uint4*)(kr + 32);
      b1.q[1] = *(const uint4*)(kr + 40);
      if (jt + 8 <= s)  // CDNA5 global_prefetch_b8 on next K tile
        __builtin_prefetch(kp + (size_t)((jt + 8) * 16 + m) * 2048 + bh * 64, 0, 0);
      v8f c = vzero;
      c = __builtin_amdgcn_wmma_f32_16x16x32_bf16(false, aq0.v, false, b0.v, (short)0, c, false, false);
      c = __builtin_amdgcn_wmma_f32_16x16x32_bf16(false, aq1.v, false, b1.v, (short)0, c, false, false);
      #pragma unroll
      for (int r = 0; r < 8; ++r) {
        int mm = r + half * 8;
        int jg = jt * 16 + m;
        float v = c[r] * 0.03125f;              // scale = sqrt(1/1024) = 1/32
        if (jg > s * 16 + mm) v = -1.0e9f;      // causal mask (diag tile only)
        P[mm * 2048 + jg] = v;
      }
    } else {
      #pragma unroll
      for (int r = 0; r < 8; ++r) P[(r + half * 8) * 2048 + jt * 16 + m] = 0.0f;
    }
  }
  __syncthreads();

  // ---- softmax over columns [0, (s+1)*16); 16 threads per row ----
  {
    int row = tid >> 4, l = tid & 15;
    int cols = (s + 1) * 16;
    float mx = -3.0e38f;
    for (int j = l; j < cols; j += 16) mx = fmaxf(mx, P[row * 2048 + j]);
    red[row * 16 + l] = mx;
    __syncthreads();
    mx = red[row * 16];
    #pragma unroll
    for (int t = 1; t < 16; ++t) mx = fmaxf(mx, red[row * 16 + t]);
    __syncthreads();
    float sum = 0.f;
    for (int j = l; j < cols; j += 16) {
      float e = __expf(P[row * 2048 + j] - mx);   // masked -> exp underflows to 0
      P[row * 2048 + j] = e;
      sum += e;
    }
    red[row * 16 + l] = sum;
    __syncthreads();
    float tot = 0.f;
    #pragma unroll
    for (int t = 0; t < 16; ++t) tot += red[row * 16 + t];
    float inv = 1.0f / tot;
    for (int j = l; j < cols; j += 16) P[row * 2048 + j] *= inv;
  }
  __syncthreads();

  // ---- write attn_weights strip (16 x 2048 fp32, coalesced float4) ----
  {
    float* dst = aw + ((size_t)(bh * 2048 + s * 16)) * 2048;
    #pragma unroll
    for (int i = 0; i < 32; ++i) {
      int f4 = tid + 256 * i;
      int rr = f4 >> 9, c4 = f4 & 511;
      *(float4*)(dst + (size_t)rr * 2048 + c4 * 4) = *(const float4*)(P + rr * 2048 + c4 * 4);
    }
  }

  // ---- O = P(16x2048) @ V(2048x64): K split across 8 waves ----
  v8f o[4];
  #pragma unroll
  for (int t = 0; t < 4; ++t) o[t] = vzero;
  int bmax = (s * 16 + 15) >> 5;          // last K-block with nonzero probs
  for (int b = wave; b <= bmax; b += 8) {
    FragB pa;                              // A frag from LDS probs, fp32->bf16
    const float* pr = P + m * 2048 + b * 32;
    #pragma unroll
    for (int e = 0; e < 8; ++e) pa.h[e]     = (__bf16)pr[half * 16 + e];
    #pragma unroll
    for (int e = 0; e < 8; ++e) pa.h[8 + e] = (__bf16)pr[half * 16 + 8 + e];
    #pragma unroll
    for (int t = 0; t < 4; ++t) {
      FragB vb;                            // vT pre-transposed: contiguous per lane
      const __bf16* vr = vT + (size_t)(bh * 64 + t * 16 + m) * 2048 + b * 32 + half * 16;
      vb.q[0] = *(const uint4*)(vr);
      vb.q[1] = *(const uint4*)(vr + 8);
      o[t] = __builtin_amdgcn_wmma_f32_16x16x32_bf16(false, pa.v, false, vb.v, (short)0, o[t], false, false);
    }
  }
  __syncthreads();                         // all waves done reading P

  // ---- cross-wave reduction of partial O (reuse P as 8 x 16 x 64 fp32) ----
  #pragma unroll
  for (int t = 0; t < 4; ++t) {
    #pragma unroll
    for (int r = 0; r < 8; ++r)
      P[wave * 1024 + (r + half * 8) * 64 + t * 16 + m] = o[t][r];
  }
  __syncthreads();
  {
    int e0 = tid * 4;
    float4 acc = *(const float4*)(P + e0);
    #pragma unroll
    for (int w = 1; w < 8; ++w) {
      float4 p = *(const float4*)(P + w * 1024 + e0);
      acc.x += p.x; acc.y += p.y; acc.z += p.z; acc.w += p.w;
    }
    int mm = e0 >> 6, d0 = e0 & 63;
    union { unsigned long long u; __bf16 h[4]; } pack;
    pack.h[0] = (__bf16)acc.x; pack.h[1] = (__bf16)acc.y;
    pack.h[2] = (__bf16)acc.z; pack.h[3] = (__bf16)acc.w;
    // X2 layout = (S, B*H, Hd) flattened == rows of the final 4096x1024 GEMM
    *(unsigned long long*)(x2 + (size_t)(s * 16 + mm) * 2048 + bh * 64 + d0) = pack.u;
  }
}

// ---------------------------------------------------------------------------
// Host launch
// ---------------------------------------------------------------------------
extern "C" void kernel_launch(void* const* d_in, const int* in_sizes, int n_in,
                              void* d_out, int out_size, void* d_ws, size_t ws_size,
                              hipStream_t stream)
{
  (void)in_sizes; (void)n_in; (void)out_size; (void)ws_size;

  const float* q  = (const float*)d_in[0];
  const float* k  = (const float*)d_in[1];
  const float* v  = (const float*)d_in[2];
  const float* Wq = (const float*)d_in[3];
  const float* bq = (const float*)d_in[4];
  const float* Wk = (const float*)d_in[5];
  const float* bk = (const float*)d_in[6];
  const float* Wv = (const float*)d_in[7];
  const float* bv = (const float*)d_in[8];
  const float* Wo = (const float*)d_in[9];
  const float* bo = (const float*)d_in[10];

  // Workspace layout (64 MiB total)
  char* ws = (char*)d_ws;
  __bf16* qbf = (__bf16*)(ws + 0);
  __bf16* kbf = (__bf16*)(ws + 8388608);
  __bf16* vbf = (__bf16*)(ws + 16777216);
  __bf16* wqb = (__bf16*)(ws + 25165824);
  __bf16* wkb = (__bf16*)(ws + 27262976);
  __bf16* wvb = (__bf16*)(ws + 29360128);
  __bf16* wob = (__bf16*)(ws + 31457280);
  __bf16* qpb = (__bf16*)(ws + 33554432);  // Q projection, bf16 row-major
  __bf16* kpb = (__bf16*)(ws + 41943040);  // K projection, bf16 row-major
  __bf16* vTb = (__bf16*)(ws + 50331648);  // V projection, per-head transposed
  __bf16* x2b = (__bf16*)(ws + 58720256);  // attention output (final GEMM input)

  float* outp = (float*)d_out;
  float* awp  = outp + NQ;                 // attn_weights region (32*2048*2048)

  cvt_f32_to_bf16<<<NQ / 1024, 256, 0, stream>>>(q,  qbf, (int)NQ);
  cvt_f32_to_bf16<<<NQ / 1024, 256, 0, stream>>>(k,  kbf, (int)NQ);
  cvt_f32_to_bf16<<<NQ / 1024, 256, 0, stream>>>(v,  vbf, (int)NQ);
  cvt_f32_to_bf16<<<NW / 1024, 256, 0, stream>>>(Wq, wqb, (int)NW);
  cvt_f32_to_bf16<<<NW / 1024, 256, 0, stream>>>(Wk, wkb, (int)NW);
  cvt_f32_to_bf16<<<NW / 1024, 256, 0, stream>>>(Wv, wvb, (int)NW);
  cvt_f32_to_bf16<<<NW / 1024, 256, 0, stream>>>(Wo, wob, (int)NW);

  dim3 gg(16, 64);   // N-tiles x M-tiles of 64
  gemm_relu_bf16<<<gg, 256, 0, stream>>>(qbf, wqb, bq, nullptr, qpb, 0);
  gemm_relu_bf16<<<gg, 256, 0, stream>>>(kbf, wkb, bk, nullptr, kpb, 0);
  gemm_relu_bf16<<<gg, 256, 0, stream>>>(vbf, wvb, bv, nullptr, vTb, 1);

  (void)hipFuncSetAttribute((const void*)attn_strip,
                            hipFuncAttributeMaxDynamicSharedMemorySize, ATTN_SMEM);
  attn_strip<<<dim3(128, 32), 256, ATTN_SMEM, stream>>>(qpb, kpb, vTb, awp, x2b);

  gemm_relu_bf16<<<gg, 256, 0, stream>>>(x2b, wob, bo, outp, nullptr, 2);
}
